// MoELayer_4183298146729
// MI455X (gfx1250) — matrix operand order repaired and checked by reading, not compiled, not run on hardware
//
#include <hip/hip_runtime.h>
#include <hip/hip_bf16.h>
#include <math.h>

// ---------------- problem constants ----------------
constexpr int B_  = 2048;   // tokens
constexpr int DR  = 1024;   // router in dim
constexpr int DI  = 1024;   // expert in dim
constexpr int DO  = 1024;   // expert out dim
constexpr int E   = 8;      // experts
constexpr int K_  = 2;      // top-k
constexpr int H1  = 2048;
constexpr int H2  = 2048;
constexpr int SLOTS  = B_ * K_;     // 4096 token-slots
constexpr int ROWS   = 4608;        // rows padded per-expert to 64: 4096 + 8*64
constexpr int MTILES = ROWS / 64;   // 72 M-tiles of 64 rows

// ---------------- workspace layout (bytes), total ~132 MB ----------------
constexpr size_t WS_GW     = 0;           // 4096 f32 gating weights (per slot)
constexpr size_t WS_PBASE  = 16384;       // 9  i32 padded expert bases
constexpr size_t WS_COUNTS = 16448;       // 8  i32
constexpr size_t WS_PERM   = 16512;       // 4608 i32 (row -> slot, -1 = pad)
constexpr size_t WS_XG     = 65536;       // 4608*1024 bf16 gathered activations
constexpr size_t WS_H1     = 16777216;    // 4608*2048 bf16
constexpr size_t WS_H2     = 35651584;    // 4608*2048 bf16
constexpr size_t WS_OUTS   = 54525952;    // 4096*1024 f32 per-slot outputs
constexpr size_t WS_WT     = 71303168;    // 64 MB bf16 K-major weights, reused per layer

typedef __attribute__((ext_vector_type(16))) __bf16 v16bf;
typedef __attribute__((ext_vector_type(8)))  float  v8f;

__device__ __forceinline__ unsigned short f2bf(float f) {
    unsigned u = __builtin_bit_cast(unsigned, f);
    u += 0x7fffu + ((u >> 16) & 1u);           // round-to-nearest-even
    return (unsigned short)(u >> 16);
}

// ---------------- 1) router: one wave per token ----------------
__global__ __launch_bounds__(256) void k_router(
    const float* __restrict__ xr, const float* __restrict__ Wg,
    const float* __restrict__ bg, const float* __restrict__ eb,
    float* __restrict__ gw, int* __restrict__ idx_out)
{
    int lane = threadIdx.x & 31;
    int wv   = threadIdx.x >> 5;
    int t    = blockIdx.x * 8 + wv;
    if (t >= B_) return;

    float acc[E];
#pragma unroll
    for (int e = 0; e < E; ++e) acc[e] = 0.f;

    const float* xrow = xr + (size_t)t * DR;
    for (int k = lane; k < DR; k += 32) {
        float xv = xrow[k];
        const float4* wrow = (const float4*)(Wg + (size_t)k * E);
        float4 w0 = wrow[0], w1 = wrow[1];
        acc[0] += xv * w0.x; acc[1] += xv * w0.y;
        acc[2] += xv * w0.z; acc[3] += xv * w0.w;
        acc[4] += xv * w1.x; acc[5] += xv * w1.y;
        acc[6] += xv * w1.z; acc[7] += xv * w1.w;
    }
#pragma unroll
    for (int e = 0; e < E; ++e)
#pragma unroll
        for (int off = 16; off > 0; off >>= 1)
            acc[e] += __shfl_xor(acc[e], off, 32);

    if (lane == 0) {
        float gate[E], sc[E];
#pragma unroll
        for (int e = 0; e < E; ++e) { gate[e] = acc[e] + bg[e]; sc[e] = gate[e] + eb[e]; }
        int e1 = -1, e2 = -1; float s1 = -1e30f, s2 = -1e30f;
#pragma unroll
        for (int e = 0; e < E; ++e) {
            float s = sc[e];
            if (s > s1) { s2 = s1; e2 = e1; s1 = s; e1 = e; }
            else if (s > s2) { s2 = s; e2 = e; }
        }
        float r1 = gate[e1], r2 = gate[e2];
        float m  = fmaxf(r1, r2);
        float w1 = __expf(r1 - m), w2 = __expf(r2 - m);
        float inv = 1.f / (w1 + w2);
        gw[t * 2 + 0] = w1 * inv;  gw[t * 2 + 1] = w2 * inv;
        idx_out[t * 2 + 0] = e1;   idx_out[t * 2 + 1] = e2;
    }
}

// ---------------- 2) deterministic bucketing (single block) ----------------
__global__ __launch_bounds__(256) void k_bucket(
    const int* __restrict__ idx, int* __restrict__ perm,
    int* __restrict__ counts, int* __restrict__ pbase)
{
    int tid = threadIdx.x, lane = tid & 31, wv = tid >> 5;  // wave wv owns expert wv
    for (int p = tid; p < ROWS; p += 256) perm[p] = -1;
    __syncthreads();

    int cnt = 0;
    for (int n0 = 0; n0 < SLOTS; n0 += 32) {
        int my = idx[n0 + lane];
        unsigned bal = (unsigned)__ballot(my == wv);
        cnt += __popc(bal);
    }
    if (lane == 0) counts[wv] = cnt;
    __syncthreads();

    if (tid == 0) {
        int b = 0;
        for (int e = 0; e < E; ++e) { pbase[e] = b; b += ((counts[e] + 63) & ~63); }
        pbase[E] = b;
    }
    __syncthreads();

    int run = pbase[wv];
    for (int n0 = 0; n0 < SLOTS; n0 += 32) {
        int n = n0 + lane;
        bool match = (idx[n] == wv);
        unsigned bal = (unsigned)__ballot(match);
        if (match) perm[run + __popc(bal & ((1u << lane) - 1u))] = n;
        run += __popc(bal);
    }
}

// ---------------- 3) gather + fp32 -> bf16 convert ----------------
__global__ __launch_bounds__(256) void k_gather(
    const float* __restrict__ xe, const int* __restrict__ perm,
    unsigned short* __restrict__ Xg)
{
    int p = blockIdx.x;
    int slot = perm[p];
    unsigned short* dst = Xg + (size_t)p * DI;
    if (slot >= 0) {
        const float* src = xe + (size_t)(slot >> 1) * DI;   // x_flat = repeat(x, K)
        for (int k = threadIdx.x * 4; k < DI; k += 256 * 4) {
            float4 v = *(const float4*)(src + k);
            uint2 pk;
            pk.x = (unsigned)f2bf(v.x) | ((unsigned)f2bf(v.y) << 16);
            pk.y = (unsigned)f2bf(v.z) | ((unsigned)f2bf(v.w) << 16);
            *(uint2*)(dst + k) = pk;
        }
    } else {
        for (int k = threadIdx.x * 4; k < DI; k += 256 * 4)
            *(uint2*)(dst + k) = make_uint2(0u, 0u);
    }
}

// ---------------- 3b) weight convert+transpose: fp32 [E][Kd][Nd] -> bf16 [E][Nd][Kd]
__global__ __launch_bounds__(256) void k_convw(
    const float* __restrict__ W, unsigned short* __restrict__ Wt,
    int Kd, int Nd)
{
    int e  = blockIdx.z;
    int k0 = blockIdx.x * 64;
    int n0 = blockIdx.y * 64;
    const float*    Wsrc = W  + (size_t)e * Kd * Nd;
    unsigned short* Wdst = Wt + (size_t)e * Nd * Kd;

    __shared__ __align__(16) unsigned short sT[64 * 72];   // [n][k], +8 pad

    int tid = threadIdx.x;
    int nl = tid & 63;
    int kb = tid >> 6;          // 0..3
#pragma unroll
    for (int r = 0; r < 16; ++r) {
        int kl = kb + r * 4;    // coalesced fp32 reads along n
        sT[nl * 72 + kl] = f2bf(Wsrc[(size_t)(k0 + kl) * Nd + n0 + nl]);
    }
    __syncthreads();
    int nw = tid >> 2;          // 0..63
    int kc = (tid & 3) * 16;    // 0,16,32,48
    const uint4* srcq = (const uint4*)(&sT[nw * 72 + kc]);
    uint4* dstq = (uint4*)(Wdst + (size_t)(n0 + nw) * Kd + k0 + kc);
    dstq[0] = srcq[0];          // 128B-contiguous bf16 writes per 4 threads
    dstq[1] = srcq[1];
}

// ---------------- 4) grouped GEMM: 64x128 block tile, bf16 WMMA ----------------
// mode 0/1: Hout = relu(A @ W[e] + b[e]) as bf16
// mode 2  : outs[slot] = gw[slot] * (A @ W[e] + b[e])    (per-slot, no atomics)
__global__ __launch_bounds__(128) void k_gemm(
    const unsigned short* __restrict__ A,   // [ROWS][Kd] bf16
    const unsigned short* __restrict__ Wt,  // [E][Nd][Kd] bf16 (K-major)
    const float* __restrict__ bias,         // [E][Nd] f32
    const int* __restrict__ pbase,          // [E+1], 64-aligned
    unsigned short* __restrict__ Hout,
    float* __restrict__ outs,
    const int* __restrict__ perm,
    const float* __restrict__ gw,
    int Kd, int Nd, int mode)
{
    int row0 = blockIdx.y * 64;
    if (row0 >= pbase[E]) return;           // past last (padded) expert range
    int e = 0;
#pragma unroll
    for (int i = 0; i < E; ++i) if (row0 >= pbase[i + 1]) e = i + 1;

    int n0 = blockIdx.x * 128;
    const float* be = bias + (size_t)e * Nd;

    __shared__ __align__(16) unsigned short sA[64 * 40];    // 80B row stride, 5 KB
    __shared__ __align__(16) unsigned short sB[128 * 40];   // [n][k], 10 KB

    int tid = threadIdx.x, lane = tid & 31, wv = tid >> 5;

    v8f acc0[4], acc1[4];
#pragma unroll
    for (int mt = 0; mt < 4; ++mt) { acc0[mt] = {}; acc1[mt] = {}; }

    // staging maps
    int am  = tid >> 1;            // A row 0..63 (2 threads / row)
    int ak  = (tid & 1) * 16;      // A k offset 0 or 16 (16 bf16 = 32B)
    const unsigned short* Arow = A  + (size_t)(row0 + am) * Kd + ak;
    const unsigned short* Wcol = Wt + (size_t)e * Nd * Kd
                                    + (size_t)(n0 + tid) * Kd;   // this thread's column

    for (int k0 = 0; k0 < Kd; k0 += 32) {
        // ---- stage A tile: 64x32 bf16, one 16B load+store per thread ----
        *(uint4*)(&sA[am * 40 + ak]) = *(const uint4*)(Arow + k0);
        // ---- stage B tile: straight copy, 64B per thread (own column, K-major) ----
        {
            const uint4* src = (const uint4*)(Wcol + k0);
            uint4* dst = (uint4*)(&sB[tid * 40]);
            dst[0] = src[0]; dst[1] = src[1]; dst[2] = src[2]; dst[3] = src[3];
        }
        __syncthreads();

        // ---- fragments per the 16-bit WMMA VGPR layouts (two b128 each) ----
        union Frag { v16bf v; uint4 q[2]; };
        Frag af[4], b0, b1;
        {
            int m = lane & 15, hi = lane >> 4;
            // A: lanes 0-15 K={0..7,16..23}; lanes 16-31 K={8..15,24..31}
#pragma unroll
            for (int mt = 0; mt < 4; ++mt) {
                const unsigned short* ab = &sA[(mt * 16 + m) * 40 + hi * 8];
                af[mt].q[0] = *(const uint4*)ab;
                af[mt].q[1] = *(const uint4*)(ab + 16);
            }
            // B: lanes 0-15 K={0..7,16..23}; lanes 16-31 K={8..15,24..31} (column = lane&15)
            int nc0 = wv * 32 + m, nc1 = nc0 + 16;
            b0.q[0] = *(const uint4*)(&sB[nc0 * 40 + hi * 8]);
            b0.q[1] = *(const uint4*)(&sB[nc0 * 40 + hi * 8 + 16]);
            b1.q[0] = *(const uint4*)(&sB[nc1 * 40 + hi * 8]);
            b1.q[1] = *(const uint4*)(&sB[nc1 * 40 + hi * 8 + 16]);
        }
#pragma unroll
        for (int mt = 0; mt < 4; ++mt) {
            acc0[mt] = __builtin_amdgcn_wmma_f32_16x16x32_bf16(
                false, af[mt].v, false, b0.v, (short)0, acc0[mt], false, false);
            acc1[mt] = __builtin_amdgcn_wmma_f32_16x16x32_bf16(
                false, af[mt].v, false, b1.v, (short)0, acc1[mt], false, false);
        }
        __syncthreads();

        if (k0 + 32 < Kd)   // prefetch next K slice of this thread's weight column
            __builtin_prefetch(Wcol + k0 + 32, 0, 1);
    }

    // ---- epilogue; C layout: VGPR r, lanes<16 -> M=r, lanes>=16 -> M=r+8 ----
    int hi = lane >> 4;
#pragma unroll
    for (int s = 0; s < 2; ++s) {
        int gcol = n0 + wv * 32 + s * 16 + (lane & 15);
        float bv = be[gcol];
#pragma unroll
        for (int mt = 0; mt < 4; ++mt) {
            v8f cc;
            if (s == 0) cc = acc0[mt]; else cc = acc1[mt];
#pragma unroll
            for (int r = 0; r < 8; ++r) {
                int p = row0 + mt * 16 + r + hi * 8;
                float val = cc[r] + bv;
                if (mode < 2) {
                    Hout[(size_t)p * Nd + gcol] = f2bf(fmaxf(val, 0.f));
                } else {
                    int slot = perm[p];
                    if (slot >= 0)
                        outs[(size_t)slot * DO + gcol] = gw[slot] * val;
                }
            }
        }
    }
}

// ---------------- 5) combine K=2 slots per token (deterministic) ----------------
__global__ __launch_bounds__(256) void k_combine(
    const float* __restrict__ outs, float* __restrict__ out)
{
    int t = blockIdx.x;
    const float4* a = (const float4*)(outs + (size_t)(2 * t) * DO);
    const float4* b = a + DO / 4;           // slot 2t+1 row is adjacent
    float4* o = (float4*)(out + (size_t)t * DO);
    for (int i = threadIdx.x; i < DO / 4; i += 256) {
        float4 x = a[i], y = b[i];
        o[i] = make_float4(x.x + y.x, x.y + y.y, x.z + y.z, x.w + y.w);
    }
}

// ---------------- host launcher ----------------
extern "C" void kernel_launch(void* const* d_in, const int* in_sizes, int n_in,
                              void* d_out, int out_size, void* d_ws, size_t ws_size,
                              hipStream_t stream) {
    (void)in_sizes; (void)n_in; (void)out_size; (void)ws_size;
    const float* xr = (const float*)d_in[0];
    const float* xe = (const float*)d_in[1];
    const float* Wg = (const float*)d_in[2];
    const float* bg = (const float*)d_in[3];
    const float* eb = (const float*)d_in[4];
    const float* W1 = (const float*)d_in[5];
    const float* b1 = (const float*)d_in[6];
    const float* W2 = (const float*)d_in[7];
    const float* b2 = (const float*)d_in[8];
    const float* W3 = (const float*)d_in[9];
    const float* b3 = (const float*)d_in[10];

    float* out_final = (float*)d_out;                    // [B, DO] f32
    int*   out_idx   = (int*)d_out + (size_t)B_ * DO;    // [B, K] i32 (bit-stored)

    char* ws = (char*)d_ws;   // requires ws_size >= ~132 MB
    float*          gw     = (float*)(ws + WS_GW);
    int*            pbase  = (int*)(ws + WS_PBASE);
    int*            counts = (int*)(ws + WS_COUNTS);
    int*            perm   = (int*)(ws + WS_PERM);
    unsigned short* Xg     = (unsigned short*)(ws + WS_XG);
    unsigned short* h1     = (unsigned short*)(ws + WS_H1);
    unsigned short* h2     = (unsigned short*)(ws + WS_H2);
    float*          outs   = (float*)(ws + WS_OUTS);
    unsigned short* Wtb    = (unsigned short*)(ws + WS_WT);  // reused per layer

    k_router <<<B_ / 8, 256, 0, stream>>>(xr, Wg, bg, eb, gw, out_idx);
    k_bucket <<<1,      256, 0, stream>>>(out_idx, perm, counts, pbase);
    k_gather <<<ROWS,   256, 0, stream>>>(xe, perm, Xg);

    // layer 1
    k_convw  <<<dim3(DI / 64, H1 / 64, E), 256, 0, stream>>>(W1, Wtb, DI, H1);
    k_gemm   <<<dim3(H1 / 128, MTILES), 128, 0, stream>>>(Xg, Wtb, b1, pbase, h1,
                                                          nullptr, perm, gw, DI, H1, 0);
    // layer 2
    k_convw  <<<dim3(H1 / 64, H2 / 64, E), 256, 0, stream>>>(W2, Wtb, H1, H2);
    k_gemm   <<<dim3(H2 / 128, MTILES), 128, 0, stream>>>(h1, Wtb, b2, pbase, h2,
                                                          nullptr, perm, gw, H1, H2, 1);
    // layer 3
    k_convw  <<<dim3(H2 / 64, DO / 64, E), 256, 0, stream>>>(W3, Wtb, H2, DO);
    k_gemm   <<<dim3(DO / 128, MTILES), 128, 0, stream>>>(h2, Wtb, b3, pbase, nullptr,
                                                          outs, perm, gw, H2, DO, 2);

    k_combine<<<B_,     256, 0, stream>>>(outs, out_final);
}
